// TemporalMatchPredictor_15393162788899
// MI455X (gfx1250) — matrix operand order repaired and checked by reading, not compiled, not run on hardware
//
#include <hip/hip_runtime.h>
#include <hip/hip_bf16.h>

#define HIDD 128
#define BN 64

typedef __attribute__((ext_vector_type(16))) __bf16 v16bf;
typedef __attribute__((ext_vector_type(8)))  __bf16 v8bf;
typedef __attribute__((ext_vector_type(8)))  float  v8f;

// ---------- helpers ----------
__device__ __forceinline__ float leaky02(float e) { return e > 0.f ? e : 0.2f * e; }
__device__ __forceinline__ float elu1(float v)    { return v > 0.f ? v : (__expf(v) - 1.f); }
// order-preserving float<->uint map for atomicMax on floats
__device__ __forceinline__ unsigned enc_f(float f) {
    unsigned u = __float_as_uint(f);
    return (u & 0x80000000u) ? ~u : (u | 0x80000000u);
}
__device__ __forceinline__ float dec_f(unsigned u) {
    u = (u & 0x80000000u) ? (u & 0x7fffffffu) : ~u;
    return __uint_as_float(u);
}

// ---------- misc init ----------
__global__ void k_zero_pool(float* pool, float* cnt, int np, int nc) {
    int i = blockIdx.x * blockDim.x + threadIdx.x;
    if (i < np) pool[i] = 0.f;
    if (i < nc) cnt[i] = 0.f;
}

// Pack a 128x128 f32 weight (row-major, K x N) into WMMA-B-fragment-ready bf16:
// packed[((c*4+kt)*32 + lane)*16 + j] = W[(kt*32 + (lane<16?0:16) + j)*128 + c*16 + (lane&15)]
__global__ void k_pack_w(const float* __restrict__ W, __bf16* __restrict__ Wp) {
    int idx = blockIdx.x * blockDim.x + threadIdx.x;   // 0..16383
    if (idx >= 8 * 4 * 32 * 16) return;
    int j    = idx & 15;
    int lane = (idx >> 4) & 31;
    int kt   = (idx >> 9) & 3;
    int c    = idx >> 11;
    int K = kt * 32 + ((lane < 16) ? 0 : 16) + j;
    int n = c * 16 + (lane & 15);
    Wp[idx] = (__bf16)W[K * HIDD + n];
}

// ---------- layer 1: thin GEMM (K=6/7) fused with attention dot products ----------
__global__ __launch_bounds__(128) void k_gat_layer1(
    const float* __restrict__ X, const float* __restrict__ W1,
    const float* __restrict__ asrc, const float* __restrict__ adst,
    float* __restrict__ H, float* __restrict__ es, float* __restrict__ ed, int Kin) {
    __shared__ float xs[8];
    __shared__ float r1[128], r2[128];
    int n = blockIdx.x, j = threadIdx.x;
    if (j < Kin) xs[j] = X[n * Kin + j];
    __syncthreads();
    float acc = 0.f;
    for (int k = 0; k < Kin; ++k) acc += xs[k] * W1[k * HIDD + j];
    H[n * HIDD + j] = acc;
    r1[j] = acc * asrc[j];
    r2[j] = acc * adst[j];
    __syncthreads();
    for (int s = 64; s > 0; s >>= 1) {
        if (j < s) { r1[j] += r1[j + s]; r2[j] += r2[j + s]; }
        __syncthreads();
    }
    if (j == 0) { es[n] = r1[0]; ed[n] = r2[0]; }
}

// ---------- layers 2/3: bf16 WMMA GEMM  H[N x128] = Xc[N x128] @ W[128x128],
// fused with attention dot products (es = H.asrc, ed = H.adst), coalesced stores.
__global__ __launch_bounds__(256) void k_gemm_bf16_wmma(
    const float* __restrict__ X, const __bf16* __restrict__ Wp,
    float* __restrict__ Hout,
    const float* __restrict__ asrc, const float* __restrict__ adst,
    float* __restrict__ es, float* __restrict__ ed, int nrows) {
    __shared__ __bf16 As[16 * 136];   // 16 rows x 128 K (bf16), +8 pad
    __shared__ float  Cs[16 * 132];   // 16 rows x 128 out (f32), +4 pad
    __shared__ float  rs[16][17], rd[16][17];
    const int rowBase = blockIdx.x * 16;
    const int t = threadIdx.x;
    // cooperative load + f32->bf16 convert: each thread 8 consecutive elements
    {
        int r = t >> 4;            // 0..15
        int c = (t & 15) * 8;      // 0..120
        __bf16* dstp = As + r * 136 + c;
        if (rowBase + r < nrows) {
            const float* srcp = X + (size_t)(rowBase + r) * HIDD + c;
#pragma unroll
            for (int i = 0; i < 8; ++i) dstp[i] = (__bf16)srcp[i];
        } else {
#pragma unroll
            for (int i = 0; i < 8; ++i) dstp[i] = (__bf16)0.f;
        }
    }
    __syncthreads();

    const int wave = t >> 5;        // col tile 0..7
    const int lane = t & 31;
    {
        const int mrow = lane & 15;
        const int abase = (lane < 16) ? 0 : 16;
        v8f c = {};
        const __bf16* wp = Wp + ((size_t)(wave * 4) * 32 + lane) * 16;
#pragma unroll
        for (int kt = 0; kt < 4; ++kt) {
            union { v16bf v; v8bf h2[2]; } a;
            const __bf16* ap = As + mrow * 136 + kt * 32 + abase;
            a.h2[0] = *(const v8bf*)(ap);        // K = kt*32 + base + 0..7
            a.h2[1] = *(const v8bf*)(ap + 8);    // K = kt*32 + base + 8..15
            v16bf b = *(const v16bf*)(wp + (size_t)kt * 32 * 16);
            c = __builtin_amdgcn_wmma_f32_16x16x32_bf16(false, a.v, false, b,
                                                        (short)0, c, false, false);
        }
        // frag -> LDS tile: lane(l) vgpr(r) -> (M = r + (l<16?0:8), N = l&15)
        const int colBase = wave * 16;
        const int n = lane & 15;
        const int mAdd = (lane < 16) ? 0 : 8;
#pragma unroll
        for (int r = 0; r < 8; ++r)
            Cs[(r + mAdd) * 132 + colBase + n] = c[r];
    }
    __syncthreads();
    // coalesced b128 stores + attention dot partials from the LDS tile
    {
        int r = t >> 4;            // row 0..15
        int p = t & 15;            // 16 partials per row
        int cc = p * 8;
        float ps = 0.f, pd = 0.f;
        if (rowBase + r < nrows) {
            const float* cp = Cs + r * 132 + cc;
            float4 v0 = *(const float4*)(cp);
            float4 v1 = *(const float4*)(cp + 4);
            float* op = Hout + (size_t)(rowBase + r) * HIDD + cc;
            *(float4*)(op)     = v0;
            *(float4*)(op + 4) = v1;
#pragma unroll
            for (int i = 0; i < 8; ++i) {
                float h = cp[i];
                ps += h * asrc[cc + i];
                pd += h * adst[cc + i];
            }
        }
        rs[r][p] = ps;
        rd[r][p] = pd;
    }
    __syncthreads();
    if (t < 16 && rowBase + t < nrows) {
        float s1 = 0.f, s2 = 0.f;
#pragma unroll
        for (int i = 0; i < 16; ++i) { s1 += rs[t][i]; s2 += rd[t][i]; }
        es[rowBase + t] = s1;
        ed[rowBase + t] = s2;
    }
}

// ---------- segment softmax over incoming edges (self-loops folded in) ----------
__global__ void k_self_max_init(const float* es, const float* ed, unsigned* mu, int N) {
    int n = blockIdx.x * blockDim.x + threadIdx.x;
    if (n < N) mu[n] = enc_f(leaky02(es[n] + ed[n]));
}
__global__ void k_edge_max(const int* __restrict__ src, const int* __restrict__ dst,
                           const float* __restrict__ es, const float* __restrict__ ed,
                           unsigned* mu, int E) {
    int i = blockIdx.x * blockDim.x + threadIdx.x;
    if (i >= E) return;
    int s = src[i], d = dst[i];
    atomicMax(&mu[d], enc_f(leaky02(es[s] + ed[d])));
}
__global__ void k_denom_init(const float* es, const float* ed, const unsigned* mu,
                             float* den, int N) {
    int n = blockIdx.x * blockDim.x + threadIdx.x;
    if (n < N) den[n] = __expf(leaky02(es[n] + ed[n]) - dec_f(mu[n]));
}
__global__ void k_edge_denom(const int* __restrict__ src, const int* __restrict__ dst,
                             const float* __restrict__ es, const float* __restrict__ ed,
                             const unsigned* __restrict__ mu, float* den, int E) {
    int i = blockIdx.x * blockDim.x + threadIdx.x;
    if (i >= E) return;
    int s = src[i], d = dst[i];
    atomicAdd(&den[d], __expf(leaky02(es[s] + ed[d]) - dec_f(mu[d])));
}
// agg[n][:] = alpha_self * h[n][:]
__global__ void k_agg_self(const float* __restrict__ H, const float* es, const float* ed,
                           const unsigned* mu, const float* den, float* agg, int N) {
    int tid = blockIdx.x * blockDim.x + threadIdx.x;
    if (tid >= N * HIDD) return;
    int n = tid >> 7;
    float alpha = __expf(leaky02(es[n] + ed[n]) - dec_f(mu[n])) / den[n];
    agg[tid] = alpha * H[tid];
}
// one wave per edge: agg[d][:] += alpha * h[s][:]
__global__ __launch_bounds__(256) void k_edge_scatter(
    const int* __restrict__ src, const int* __restrict__ dst,
    const float* __restrict__ es, const float* __restrict__ ed,
    const unsigned* __restrict__ mu, const float* __restrict__ den,
    const float* __restrict__ H, float* __restrict__ agg, int E) {
    int e = blockIdx.x * 8 + (threadIdx.x >> 5);
    if (e >= E) return;
    int lane = threadIdx.x & 31;
    int s = src[e], d = dst[e];
    float ev = leaky02(es[s] + ed[d]);
    float alpha = __expf(ev - dec_f(mu[d])) / den[d];
    const float4 hv = ((const float4*)(H + (size_t)s * HIDD))[lane];
    float* ap = agg + (size_t)d * HIDD + lane * 4;
    atomicAdd(ap + 0, alpha * hv.x);
    atomicAdd(ap + 1, alpha * hv.y);
    atomicAdd(ap + 2, alpha * hv.z);
    atomicAdd(ap + 3, alpha * hv.w);
}
__global__ void k_bias_elu(const float* __restrict__ agg, const float* __restrict__ bias,
                           float* __restrict__ xc, int N) {
    int tid = blockIdx.x * blockDim.x + threadIdx.x;
    if (tid >= N * HIDD) return;
    xc[tid] = elu1(agg[tid] + bias[tid & 127]);
}

// ---------- global mean pool (sum + count; mean applied in fusion) ----------
__global__ void k_pool_add(const float* __restrict__ xc, const int* __restrict__ batch,
                           float* pool, float* cnt, int N, int slot) {
    int tid = blockIdx.x * blockDim.x + threadIdx.x;
    if (tid >= N * HIDD) return;
    int n = tid >> 7, j = tid & 127;
    int b = batch[n];
    atomicAdd(pool + ((size_t)(slot * BN + b)) * HIDD + j, xc[tid]);
    if (j == 0) atomicAdd(cnt + slot * BN + b, 1.f);
}

// ---------- fusion MLP + classifier ----------
__global__ __launch_bounds__(128) void k_fusion_clf(
    const float* __restrict__ pool, const float* __restrict__ cnt,
    const float* f0, const float* f1, const float* f2,
    const float* f3, const float* f4, const float* f5,
    const float* __restrict__ fW, const float* __restrict__ fb,
    const float* __restrict__ cW, const float* __restrict__ cb,
    float* __restrict__ out) {
    __shared__ float z[128];
    int b = blockIdx.x, j = threadIdx.x;
    float acc = fb[j];
    for (int g = 0; g < 6; ++g) {
        float cv = cnt[g * BN + b];
        cv = cv > 1.f ? cv : 1.f;
        float inv = 1.f / cv;
        const float* pr = pool + (size_t)(g * BN + b) * HIDD;
        const float* wr = fW + (size_t)(g * HIDD) * HIDD + j;
        for (int k = 0; k < HIDD; ++k) acc += (pr[k] * inv) * wr[(size_t)k * HIDD];
    }
    const float* feats[6] = {f0, f1, f2, f3, f4, f5};
    for (int t = 0; t < 6; ++t) {
        const float* fr = feats[t] + b * 22;
        const float* wr = fW + (size_t)(6 * HIDD + t * 22) * HIDD + j;
        for (int k = 0; k < 22; ++k) acc += fr[k] * wr[(size_t)k * HIDD];
    }
    z[j] = elu1(acc);
    __syncthreads();
    if (j < 3) {
        float o = cb[j];
        for (int k = 0; k < HIDD; ++k) o += z[k] * cW[k * 3 + j];
        out[b * 3 + j] = o;
    }
}

// ---------- launch ----------
extern "C" void kernel_launch(void* const* d_in, const int* in_sizes, int n_in,
                              void* d_out, int out_size, void* d_ws, size_t ws_size,
                              hipStream_t stream) {
    (void)n_in; (void)out_size; (void)ws_size;
    struct GDesc { int xi, ei, bi, kin, cum, slot; };
    // processed in input order; slot = position in the fusion concat
    const GDesc gs[6] = {
        {0, 1, 2, 7, 1, 0},   // hc
        {3, 4, 5, 7, 1, 1},   // ac
        {6, 7, 8, 6, 0, 2},   // hi0
        {9, 10, 11, 6, 0, 4}, // hi1 -> slot 4 (concat: hc,ac,hi0,ai0,hi1,ai1)
        {12, 13, 14, 6, 0, 3},// ai0 -> slot 3
        {15, 16, 17, 6, 0, 5},// ai1
    };
    const float* cumW[3] = {(const float*)d_in[24], (const float*)d_in[25], (const float*)d_in[26]};
    const float* cum_att  = (const float*)d_in[27];
    const float* cum_bias = (const float*)d_in[28];
    const float* intW[3] = {(const float*)d_in[29], (const float*)d_in[30], (const float*)d_in[31]};
    const float* int_att  = (const float*)d_in[32];
    const float* int_bias = (const float*)d_in[33];
    const float* fW = (const float*)d_in[34];
    const float* fb = (const float*)d_in[35];
    const float* cW = (const float*)d_in[36];
    const float* cb = (const float*)d_in[37];

    int maxN = 0, maxE = 0;
    for (int g = 0; g < 6; ++g) {
        int N = in_sizes[gs[g].xi] / gs[g].kin;
        int E = in_sizes[gs[g].ei] / 2;
        if (N > maxN) maxN = N;
        if (E > maxE) maxE = E;
    }

    size_t off = 0;
    auto alloc = [&](size_t bytes) -> void* {
        size_t o = (off + 255) & ~(size_t)255;
        off = o + bytes;
        return (void*)((char*)d_ws + o);
    };
    float*    h    = (float*)alloc((size_t)maxN * HIDD * 4);
    float*    xc   = (float*)alloc((size_t)maxN * HIDD * 4);
    float*    agg  = (float*)alloc((size_t)maxN * HIDD * 4);
    float*    es   = (float*)alloc((size_t)maxN * 4);
    float*    ed   = (float*)alloc((size_t)maxN * 4);
    unsigned* mu   = (unsigned*)alloc((size_t)maxN * 4);
    float*    den  = (float*)alloc((size_t)maxN * 4);
    __bf16*   Wp   = (__bf16*)alloc((size_t)4 * 16384 * 2);
    float*    pool = (float*)alloc((size_t)6 * BN * HIDD * 4);
    float*    cnt  = (float*)alloc((size_t)6 * BN * 4);

    k_zero_pool<<<(6 * BN * HIDD + 255) / 256, 256, 0, stream>>>(pool, cnt, 6 * BN * HIDD, 6 * BN);
    k_pack_w<<<64, 256, 0, stream>>>(cumW[1], Wp + 0 * 16384);
    k_pack_w<<<64, 256, 0, stream>>>(cumW[2], Wp + 1 * 16384);
    k_pack_w<<<64, 256, 0, stream>>>(intW[1], Wp + 2 * 16384);
    k_pack_w<<<64, 256, 0, stream>>>(intW[2], Wp + 3 * 16384);

    for (int g = 0; g < 6; ++g) {
        const GDesc& G = gs[g];
        const float* X   = (const float*)d_in[G.xi];
        const int* edge  = (const int*)d_in[G.ei];
        const int* batch = (const int*)d_in[G.bi];
        int N = in_sizes[G.xi] / G.kin;
        int E = in_sizes[G.ei] / 2;
        const int* srcI = edge;
        const int* dstI = edge + E;
        const float* const* Ws = G.cum ? cumW : intW;
        const float* att  = G.cum ? cum_att : int_att;
        const float* bias = G.cum ? cum_bias : int_bias;
        const __bf16* W2p = Wp + (size_t)(G.cum ? 0 : 2) * 16384;
        const __bf16* W3p = Wp + (size_t)(G.cum ? 1 : 3) * 16384;

        for (int l = 0; l < 3; ++l) {
            const float* asrc = att + (l * 2 + 0) * HIDD;
            const float* adst = att + (l * 2 + 1) * HIDD;
            const float* bl   = bias + l * HIDD;
            if (l == 0) {
                k_gat_layer1<<<N, 128, 0, stream>>>(X, Ws[0], asrc, adst, h, es, ed, G.kin);
            } else {
                const __bf16* wpk = (l == 1) ? W2p : W3p;
                k_gemm_bf16_wmma<<<(N + 15) / 16, 256, 0, stream>>>(
                    xc, wpk, h, asrc, adst, es, ed, N);
            }
            k_self_max_init<<<(N + 255) / 256, 256, 0, stream>>>(es, ed, mu, N);
            k_edge_max<<<(E + 255) / 256, 256, 0, stream>>>(srcI, dstI, es, ed, mu, E);
            k_denom_init<<<(N + 255) / 256, 256, 0, stream>>>(es, ed, mu, den, N);
            k_edge_denom<<<(E + 255) / 256, 256, 0, stream>>>(srcI, dstI, es, ed, mu, den, E);
            k_agg_self<<<(N * HIDD + 255) / 256, 256, 0, stream>>>(h, es, ed, mu, den, agg, N);
            k_edge_scatter<<<(E + 7) / 8, 256, 0, stream>>>(srcI, dstI, es, ed, mu, den, h, agg, E);
            k_bias_elu<<<(N * HIDD + 255) / 256, 256, 0, stream>>>(agg, bl, xc, N);
        }
        k_pool_add<<<(N * HIDD + 255) / 256, 256, 0, stream>>>(xc, batch, pool, cnt, N, G.slot);
    }
    // feats concat order: hc(18), ac(19), hi0(20), ai0(22), hi1(21), ai1(23)
    k_fusion_clf<<<BN, 128, 0, stream>>>(pool, cnt,
        (const float*)d_in[18], (const float*)d_in[19], (const float*)d_in[20],
        (const float*)d_in[22], (const float*)d_in[21], (const float*)d_in[23],
        fW, fb, cW, cb, (float*)d_out);
}
